// RDECommunicationLSTM_54992761258656
// MI455X (gfx1250) — matrix operand order repaired and checked by reading, not compiled, not run on hardware
//
#include <hip/hip_runtime.h>
#include <hip/hip_bf16.h>

typedef __attribute__((ext_vector_type(16))) _Float16 v16h;
typedef __attribute__((ext_vector_type(8)))  _Float16 v8h;
typedef __attribute__((ext_vector_type(8)))  float    v8f;

#define BIN   8     // IN
#define BHID  32    // HID
#define BOUT  16    // OUT

// workspace layout (bytes):
//   [0    , 512 )  : const0[128] f32  = b_ih0+b_hh0 + h0[0] @ W_hh0^T
//   [512  , 1024)  : const1[128] f32  = b_ih1+b_hh1 + h0[1] @ W_hh1^T
//   [1024 , 19456) : 18 packed WMMA B-tiles, each 32 lanes x 16 halves
//       tiles 0..7  : W_ih0^T   (K=8 zero-padded to 32, N tile = gates 0..127)
//       tiles 8..15 : W_ih1^T   (K=32, N tile = gates 0..127)
//       tile  16    : column-centered mu_W^T (mu_W - mean_n mu_W)  (K=32, N=16)
//       tile  17    : sig_W^T    (K=32, N=16)
//   [19456, 19520) : mu_bc[16] f32 = mu_b - mean(mu_b)
#define WS_BP   1024
#define WS_MUBC (WS_BP + 18 * 1024)

__global__ void rde_setup_kernel(const float* __restrict__ W_ih0,
                                 const float* __restrict__ W_hh0,
                                 const float* __restrict__ b_ih0,
                                 const float* __restrict__ b_hh0,
                                 const float* __restrict__ W_ih1,
                                 const float* __restrict__ W_hh1,
                                 const float* __restrict__ b_ih1,
                                 const float* __restrict__ b_hh1,
                                 const float* __restrict__ h0,
                                 const float* __restrict__ mu_W,
                                 const float* __restrict__ mu_b,
                                 const float* __restrict__ sig_W,
                                 unsigned char* __restrict__ wsb) {
  const int tid = threadIdx.x;
  float* const0 = (float*)wsb;
  float* const1 = (float*)(wsb + 512);
  if (tid < 128) {
    float s = b_ih0[tid] + b_hh0[tid];
    #pragma unroll
    for (int k = 0; k < BHID; ++k) s += W_hh0[tid * BHID + k] * h0[k];
    const0[tid] = s;
  } else if (tid < 256) {
    const int nn = tid - 128;
    float s = b_ih1[nn] + b_hh1[nn];
    #pragma unroll
    for (int k = 0; k < BHID; ++k) s += W_hh1[nn * BHID + k] * h0[BHID + k];
    const1[nn] = s;
  } else if (tid < 256 + BOUT) {
    // centered mu bias
    const int nn = tid - 256;
    float m = 0.0f;
    #pragma unroll
    for (int j = 0; j < BOUT; ++j) m += mu_b[j];
    ((float*)(wsb + WS_MUBC))[nn] = mu_b[nn] - m * (1.0f / BOUT);
  }
  // pack B operands: element e of lane l holds (K = e<8 ? (l>>4)*8 + e : 16 + (l>>4)*8 + e-8, N = l&15)
  _Float16* bp = (_Float16*)(wsb + WS_BP);
  for (int q = tid; q < 18 * 512; q += blockDim.x) {
    const int tile = q >> 9;
    const int rem  = q & 511;
    const int lane = rem >> 4;
    const int e    = rem & 15;
    const int n16  = lane & 15;
    const int kk   = (e < 8) ? ((lane >> 4) << 3) + e : 16 + ((lane >> 4) << 3) + (e - 8);
    float v;
    if (tile < 8) {
      const int N = tile * 16 + n16;
      v = (kk < BIN) ? W_ih0[N * BIN + kk] : 0.0f;
    } else if (tile < 16) {
      const int N = (tile - 8) * 16 + n16;
      v = W_ih1[N * BHID + kk];
    } else if (tile == 16) {
      // column-centered mu weights: LN mean removal folded into the GEMM
      float cm = 0.0f;
      #pragma unroll
      for (int j = 0; j < BOUT; ++j) cm += mu_W[j * BHID + kk];
      v = mu_W[n16 * BHID + kk] - cm * (1.0f / BOUT);
    } else {
      v = sig_W[n16 * BHID + kk];
    }
    bp[q] = (_Float16)v;
  }
}

// fast activations on the hardware transcendental unit (v_exp/v_rcp/v_log)
__device__ __forceinline__ float fsigm(float x) {
  return __builtin_amdgcn_rcpf(1.0f + __expf(-x));
}
__device__ __forceinline__ float ftanh(float x) {
  // 1 - 2/(exp(2x)+1): exp overflow -> 1, underflow -> -1 (correct saturation)
  return 1.0f - 2.0f * __builtin_amdgcn_rcpf(__expf(2.0f * x) + 1.0f);
}

__global__ void __launch_bounds__(256)
rde_lstm_wmma_kernel(const float* __restrict__ x,
                     const float* __restrict__ c0,      // [2,1,32]
                     const float* __restrict__ ln_g,    // [16]
                     const float* __restrict__ ln_b,    // [16]
                     const float* __restrict__ sig_b,   // [16]
                     const unsigned char* __restrict__ wsb,
                     float* __restrict__ out,
                     int Btot) {
  __shared__ __align__(16) _Float16 hbuf[8 * 16 * 32];  // 8 waves x (16 rows x 32 f16)

  const int lane   = threadIdx.x & 31;
  const int w      = threadIdx.x >> 5;
  const int waveId = blockIdx.x * (blockDim.x >> 5) + w;
  const int row0   = waveId * 16;
  if (row0 >= Btot) return;  // wave-uniform

  const int n  = lane & 15;   // N / A-row index
  const int hh = lane >> 4;   // lane half

  const float* const0 = (const float*)wsb;
  const float* const1 = (const float*)(wsb + 512);
  const float* mu_bc  = (const float*)(wsb + WS_MUBC);
  const v16h*  bp     = (const v16h*)(wsb + WS_BP);

  // ---------- Layer 0 A-operand: rows of X, K padded 8 -> 32 ----------
  v16h a0;
  #pragma unroll
  for (int e = 0; e < 16; ++e) a0[e] = (_Float16)0.0f;
  if (hh == 0) {  // elements 0..7 of half-0 lanes carry K=0..7
    const float4* xr = (const float4*)(x + (size_t)(row0 + n) * BIN);
    const float4 x0 = xr[0], x1 = xr[1];
    a0[0] = (_Float16)x0.x; a0[1] = (_Float16)x0.y;
    a0[2] = (_Float16)x0.z; a0[3] = (_Float16)x0.w;
    a0[4] = (_Float16)x1.x; a0[5] = (_Float16)x1.y;
    a0[6] = (_Float16)x1.z; a0[7] = (_Float16)x1.w;
  }

  // ---------- Layer 0 gates: 8 N-tiles of 16 over 128 gate outputs ----------
  v8f g[8];
  #pragma unroll
  for (int t = 0; t < 8; ++t) {
    const float cv = const0[t * 16 + n];
    v8f acc;
    #pragma unroll
    for (int r = 0; r < 8; ++r) acc[r] = cv;
    const v16h b = bp[t * 32 + lane];
    g[t] = __builtin_amdgcn_wmma_f32_16x16x32_f16(false, a0, false, b,
                                                  (short)0, acc, false, false);
  }

  // ---------- LSTM cell 0 (c0 broadcast), h1 -> LDS (f16, row-major 16x32) ----------
  _Float16* hrow = hbuf + w * (16 * 32);
  #pragma unroll
  for (int p = 0; p < 2; ++p) {
    const int   hid = p * 16 + n;
    const float c0v = c0[hid];
    #pragma unroll
    for (int r = 0; r < 8; ++r) {
      const float iv = fsigm(g[0 + p][r]);
      const float fv = fsigm(g[2 + p][r]);
      const float gv = ftanh(g[4 + p][r]);
      const float ov = fsigm(g[6 + p][r]);
      const float c1 = fv * c0v + iv * gv;
      hrow[(r + 8 * hh) * 32 + hid] = (_Float16)(ov * ftanh(c1));
    }
  }
  asm volatile("s_wait_dscnt 0" ::: "memory");

  // ---------- reload h1 in WMMA A-layout: two contiguous 8-half runs ----------
  v16h a1;
  {
    const v8h lo = *(const v8h*)(hrow + n * 32 + hh * 8);        // K = hh*8 .. +7
    const v8h hi = *(const v8h*)(hrow + n * 32 + 16 + hh * 8);   // K = 16+hh*8 .. +7
    #pragma unroll
    for (int e = 0; e < 8; ++e) { a1[e] = lo[e]; a1[8 + e] = hi[e]; }
  }

  // ---------- Layer 1 gates ----------
  #pragma unroll
  for (int t = 0; t < 8; ++t) {
    const float cv = const1[t * 16 + n];
    v8f acc;
    #pragma unroll
    for (int r = 0; r < 8; ++r) acc[r] = cv;
    const v16h b = bp[(8 + t) * 32 + lane];
    g[t] = __builtin_amdgcn_wmma_f32_16x16x32_f16(false, a1, false, b,
                                                  (short)0, acc, false, false);
  }

  // ---------- LSTM cell 1, h2 -> LDS ----------
  #pragma unroll
  for (int p = 0; p < 2; ++p) {
    const int   hid = p * 16 + n;
    const float c0v = c0[BHID + hid];
    #pragma unroll
    for (int r = 0; r < 8; ++r) {
      const float iv = fsigm(g[0 + p][r]);
      const float fv = fsigm(g[2 + p][r]);
      const float gv = ftanh(g[4 + p][r]);
      const float ov = fsigm(g[6 + p][r]);
      const float c1 = fv * c0v + iv * gv;
      hrow[(r + 8 * hh) * 32 + hid] = (_Float16)(ov * ftanh(c1));
    }
  }
  asm volatile("s_wait_dscnt 0" ::: "memory");

  v16h a2;
  {
    const v8h lo = *(const v8h*)(hrow + n * 32 + hh * 8);
    const v8h hi = *(const v8h*)(hrow + n * 32 + 16 + hh * 8);
    #pragma unroll
    for (int e = 0; e < 8; ++e) { a2[e] = lo[e]; a2[8 + e] = hi[e]; }
  }

  // ---------- heads ----------
  // mu head uses centered weights/bias: WMMA output is already mean-removed
  v8f mu_acc, sg_acc;
  {
    const float mb = mu_bc[n], sb = sig_b[n];
    #pragma unroll
    for (int r = 0; r < 8; ++r) { mu_acc[r] = mb; sg_acc[r] = sb; }
  }
  mu_acc = __builtin_amdgcn_wmma_f32_16x16x32_f16(false, a2, false, bp[16 * 32 + lane],
                                                  (short)0, mu_acc, false, false);
  sg_acc = __builtin_amdgcn_wmma_f32_16x16x32_f16(false, a2, false, bp[17 * 32 + lane],
                                                  (short)0, sg_acc, false, false);

  const float lng = ln_g[n], lnb = ln_b[n];
  float* outs = out + (size_t)Btot * BOUT;   // sigma block of concatenated output

  #pragma unroll
  for (int r = 0; r < 8; ++r) {
    const int row = row0 + r + 8 * hh;
    // LayerNorm: d is pre-centered; only variance butterfly (within 16-lane half)
    const float d = mu_acc[r];
    float q = d * d;
    q += __shfl_xor(q, 1, 32); q += __shfl_xor(q, 2, 32);
    q += __shfl_xor(q, 4, 32); q += __shfl_xor(q, 8, 32);
    const float var = q * 0.0625f;
    out[(size_t)row * BOUT + n] = d * __builtin_amdgcn_rsqf(var + 1e-5f) * lng + lnb;
    // softplus + 1e-6
    const float xv = sg_acc[r];
    const float sp = (xv > 20.0f) ? xv : __logf(1.0f + __expf(xv));
    outs[(size_t)row * BOUT + n] = sp + 1e-6f;
  }
}

extern "C" void kernel_launch(void* const* d_in, const int* in_sizes, int n_in,
                              void* d_out, int out_size, void* d_ws, size_t ws_size,
                              hipStream_t stream) {
  (void)n_in; (void)out_size; (void)ws_size;
  const float* x     = (const float*)d_in[0];
  const float* W_ih0 = (const float*)d_in[1];
  const float* W_hh0 = (const float*)d_in[2];
  const float* b_ih0 = (const float*)d_in[3];
  const float* b_hh0 = (const float*)d_in[4];
  const float* W_ih1 = (const float*)d_in[5];
  const float* W_hh1 = (const float*)d_in[6];
  const float* b_ih1 = (const float*)d_in[7];
  const float* b_hh1 = (const float*)d_in[8];
  const float* h0    = (const float*)d_in[9];
  const float* c0    = (const float*)d_in[10];
  const float* mu_W  = (const float*)d_in[11];
  const float* mu_b  = (const float*)d_in[12];
  const float* ln_g  = (const float*)d_in[13];
  const float* ln_b  = (const float*)d_in[14];
  const float* sig_W = (const float*)d_in[15];
  const float* sig_b = (const float*)d_in[16];

  const int B = in_sizes[0] / BIN;

  rde_setup_kernel<<<1, 1024, 0, stream>>>(W_ih0, W_hh0, b_ih0, b_hh0,
                                           W_ih1, W_hh1, b_ih1, b_hh1,
                                           h0, mu_W, mu_b, sig_W,
                                           (unsigned char*)d_ws);

  const int waves  = (B + 15) / 16;
  const int blocks = (waves + 7) / 8;   // 8 waves (256 threads) per block
  rde_lstm_wmma_kernel<<<blocks, 256, 0, stream>>>(
      x, c0, ln_g, ln_b, sig_b,
      (const unsigned char*)d_ws, (float*)d_out, B);
}
